// ParallelBadgerConv_57896159150606
// MI455X (gfx1250) — matrix operand order repaired and checked by reading, not compiled, not run on hardware
//
#include <hip/hip_runtime.h>
#include <hip/hip_bf16.h>
#include <math.h>

// Sizes (fixed by the reference)
#define LSEQ 2048
#define BB   2
#define NH   8
#define HD   128
#define CCH  1024          // NH*HD
#define EMB  33
#define ORD  64
#define KREV (LSEQ + 64)   // reversed kernel row + 64-zero causal pad

typedef __bf16 v16bf __attribute__((ext_vector_type(16)));
typedef __bf16 v8bf  __attribute__((ext_vector_type(8)));
typedef float  v8f   __attribute__((ext_vector_type(8)));

// ---------------------------------------------------------------------------
// K0: hyena filter MLP -> krev[hd][x] = kern[hd][L-1-x] in bf16, with 64
// trailing zeros so the Toeplitz gather needs no causality guard at all.
// ---------------------------------------------------------------------------
__global__ void __launch_bounds__(128) hyena_filter_kernel(
    const float* __restrict__ l0w, const float* __restrict__ l0b,
    const float* __restrict__ l1w, const float* __restrict__ l1b,
    const float* __restrict__ l2w, const float* __restrict__ l2b,
    const float* __restrict__ frq, const float* __restrict__ fw,
    const float* __restrict__ mw,  __bf16* __restrict__ krevb)
{
    const int l = blockIdx.x * blockDim.x + threadIdx.x;
    if (l >= LSEQ) return;
    const float tl = (float)l / (float)(LSEQ - 1);

    float z[EMB];
    z[0] = tl;
    const float wang = 6.28318530717958647692f * (float)l / (float)LSEQ;
    #pragma unroll
    for (int j = 0; j < 16; ++j) {
        const float fj = 1e-4f + (float)j * (15.0f - 1e-4f) / 15.0f;
        const float th = wang * fj;
        z[1 + j]  =  cosf(th);
        z[17 + j] = -sinf(th);
    }

    float h[ORD], g[ORD];
    for (int o = 0; o < ORD; ++o) {
        float s = l0b[o];
        for (int e = 0; e < EMB; ++e) s += z[e] * l0w[o * EMB + e];
        h[o] = sinf(frq[o] * s);
    }
    for (int o = 0; o < ORD; ++o) {
        float s = l1b[o];
        for (int e = 0; e < ORD; ++e) s += h[e] * l1w[o * ORD + e];
        g[o] = sinf(frq[o] * s);
    }
    for (int o = 0; o < ORD; ++o) {
        float s = l2b[o];
        for (int e = 0; e < ORD; ++e) s += g[e] * l2w[o * ORD + e];
        h[o] = sinf(frq[o] * s);
    }
    const float dec = expf(-tl);   // placeholder to keep expf shared? (unused)
    (void)dec;
    for (int hd = 0; hd < HD; ++hd) {
        float s = 0.f;
        for (int o = 0; o < ORD; ++o) s += h[o] * fw[hd * ORD + o];
        krevb[(size_t)hd * KREV + (LSEQ - 1 - l)] =
            (__bf16)(s * expf(-tl * fabsf(mw[hd])));
        if (l < 64)   // zero the causal pad tail
            krevb[(size_t)hd * KREV + LSEQ + l] = (__bf16)0.0f;
    }
}

// ---------------------------------------------------------------------------
// K1: 3-tap causal depthwise conv, [L,B,NH,HD] -> [B,NH,HD,L] (LDS transpose
// so reads are hd-contiguous and writes are l-contiguous).
// ---------------------------------------------------------------------------
__global__ void __launch_bounds__(256) shortconv_kernel(
    const float* __restrict__ x,   // [L,B,NH,HD]
    const float* __restrict__ w3,  // [C][3] (this qkv slice)
    const float* __restrict__ b1,  // [C]
    float* __restrict__ y)         // [B,NH,HD,L]
{
    __shared__ float xs[34 * 33];
    __shared__ float ys[32 * 33];
    const int l0  = blockIdx.x * 32;
    const int hd0 = blockIdx.y * 32;
    const int bn  = blockIdx.z;           // b*NH+nh
    const int b   = bn >> 3, nh = bn & 7;
    const int tid = threadIdx.x;

    for (int i = tid; i < 34 * 32; i += 256) {
        const int row = i >> 5, cc = i & 31;
        const int gl = l0 + row - 2;
        float v = 0.f;
        if (gl >= 0)
            v = x[((((size_t)gl * BB + b) * NH + nh) * HD) + hd0 + cc];
        xs[row * 33 + cc] = v;
    }
    __syncthreads();
    for (int i = tid; i < 32 * 32; i += 256) {
        const int lr = i >> 5, cc = i & 31;
        const int ch = nh * HD + hd0 + cc;
        const float r = w3[ch * 3 + 0] * xs[lr * 33 + cc]
                      + w3[ch * 3 + 1] * xs[(lr + 1) * 33 + cc]
                      + w3[ch * 3 + 2] * xs[(lr + 2) * 33 + cc]
                      + b1[ch];
        ys[cc * 33 + lr] = r;
    }
    __syncthreads();
    for (int i = tid; i < 32 * 32; i += 256) {
        const int cc = i >> 5, lr = i & 31;
        y[(((size_t)(b * NH + nh)) * HD + hd0 + cc) * LSEQ + l0 + lr] = ys[cc * 33 + lr];
    }
}

// ---------------------------------------------------------------------------
// K2: kv[hd][c][l] = V[b,d1,hd,l] * K[b,d2,hd,l], c = ((b*8+d2)*8 + d1), bf16.
// ---------------------------------------------------------------------------
__global__ void __launch_bounds__(256) form_kv_kernel(
    const float* __restrict__ vc, const float* __restrict__ kc,
    __bf16* __restrict__ kvb)
{
    const int tid = blockIdx.x * 256 + threadIdx.x;   // HD*B*NH*L threads
    const int l  = tid & (LSEQ - 1);
    int rest = tid >> 11;
    const int d2 = rest & 7; rest >>= 3;
    const int b  = rest & 1;
    const int hd = rest >> 1;
    const float kval = kc[(((size_t)(b * NH + d2)) * HD + hd) * LSEQ + l];
    const size_t outbase = ((size_t)hd * 128 + (size_t)((b * NH + d2) * NH)) * LSEQ + l;
    #pragma unroll
    for (int d1 = 0; d1 < NH; ++d1) {
        const float vval = vc[(((size_t)(b * NH + d1)) * HD + hd) * LSEQ + l];
        kvb[outbase + (size_t)d1 * LSEQ] = (__bf16)(vval * kval);
    }
}

// ---------------------------------------------------------------------------
// K3: blocked lower-triangular Toeplitz matmul on the matrix cores.
//   block tile = 64 t x 128 ch for one hd; 8 waves = 4 t-rows x 2 ch-groups;
//   each wave: 1 A build -> 4 x v_wmma_f32_16x16x32_bf16 (4 accumulators).
//   Barrier/LDS-free: A operand comes from the reversed+padded filter as two
//   contiguous 8-half global loads per lane (window is L0/L2 resident).
// ---------------------------------------------------------------------------
__global__ void __launch_bounds__(256) hyena_tconv_wmma_kernel(
    const __bf16* __restrict__ krevb,  // [HD][KREV] bf16 (reversed + zero pad)
    const __bf16* __restrict__ kvb,    // [HD][128][L] bf16
    const float* __restrict__ qc,      // [B][NH][HD][L]
    const float* __restrict__ Dv,      // [HD]
    float* __restrict__ outp)          // reference output, f32
{
    const int t0   = blockIdx.x * 64;
    const int hd   = blockIdx.y;
    const int tid  = threadIdx.x;
    const int lane = tid & 31;
    const int wv   = tid >> 5;
    const int tr   = wv >> 1;          // 0..3  t-row (distinct diagonal/wave)
    const int cg   = wv & 1;           // 0..1  64-channel group
    const int col  = lane & 15;
    const int hi   = lane >> 4;        // lane half per ISA 7.12.2 layouts
    const int m    = col;              // A-matrix row (M = lane & 15)

    const __bf16* krow   = krevb + (size_t)hd * KREV;
    const size_t  kvbase = ((size_t)(hd * 128 + cg * 64) + col) * LSEQ;

    // x = (L-1) - (t - s): reversed index; causal t-s<0 lands in the zero pad
    const int xoff = (LSEQ - 1) - t0 - 16 * tr - m + 16 * hi;

    v8f acc[4] = {};

    for (int s0 = 0; s0 <= t0 + 63; s0 += 32) {
        // A operand (16-bit A 16x32): lane<16 h0..15=K0..15,
        // lane>=16 h0..15=K16..31 (matches B).  In reversed coords the
        // lane's 16 halves are one ascending contiguous run.
        const __bf16* kp = krow + (xoff + s0);
        v16bf a;
        #pragma unroll
        for (int i = 0; i < 8; ++i)  a[i] = kp[i];        // K = i + 16*hi
        #pragma unroll
        for (int i = 8; i < 16; ++i) a[i] = kp[i];        // K = i + 16*hi

        #pragma unroll
        for (int n = 0; n < 4; ++n) {
            // B operand: 32x16, lane = column, 16 contiguous K values (32B)
            const v16bf bmat =
                *(const v16bf*)(kvb + kvbase + (size_t)(16 * n) * LSEQ + s0 + hi * 16);
            acc[n] = __builtin_amdgcn_wmma_f32_16x16x32_bf16(
                false, a, false, bmat, (short)0, acc[n], false, false);
        }
        // speculative prefetch of next k-step's kv line (global_prefetch_b8)
        __builtin_prefetch((const void*)(kvb + kvbase + s0 + 32), 0, 0);
    }

    // Epilogue: out[b,d2,hd,t] = sum_d1 q[b,d1,hd,t] * (Y + D[hd]*kv)
    const float Dhd  = Dv[hd];
    const size_t trow = (size_t)t0 + 16 * tr + 8 * hi;    // 8-aligned
    #pragma unroll
    for (int n = 0; n < 4; ++n) {
        const int c   = cg * 64 + 16 * n + col;  // channel ((b*8+d2)*8 + d1)
        const int d1  = c & 7;
        const int bd2 = c >> 3;
        const int b_  = bd2 >> 3;
        const int d2  = bd2 & 7;
        const v8bf kv8 = *(const v8bf*)(kvb + ((size_t)(hd * 128 + c)) * LSEQ + trow);
        const v8f  qv  = *(const v8f*)(qc + ((((size_t)(b_ * NH + d1)) * HD + hd) * LSEQ) + trow);
        #pragma unroll
        for (int r = 0; r < 8; ++r) {            // f32 C/D: M = r + 8*hi
            float wvv = qv[r] * (acc[n][r] + Dhd * (float)kv8[r]);
            wvv += __shfl_xor(wvv, 1, 32);       // reduce 8 d1 columns
            wvv += __shfl_xor(wvv, 2, 32);
            wvv += __shfl_xor(wvv, 4, 32);
            if (d1 == 0) {
                const int cout = hd * NH + d2;   // (h d2) flat channel
                outp[((size_t)(b_ * NH + (cout >> 7)) * LSEQ + (trow + r)) * HD +
                     (cout & 127)] = wvv;
            }
        }
    }
}

// ---------------------------------------------------------------------------
extern "C" void kernel_launch(void* const* d_in, const int* in_sizes, int n_in,
                              void* d_out, int out_size, void* d_ws, size_t ws_size,
                              hipStream_t stream) {
    const float* qin = (const float*)d_in[0];
    const float* kin = (const float*)d_in[1];
    const float* vin = (const float*)d_in[2];
    const float* scw = (const float*)d_in[3];   // [3,C,1,3]
    const float* scb = (const float*)d_in[4];   // [3,C]
    const float* Dv  = (const float*)d_in[5];   // [HD]
    const float* l0w = (const float*)d_in[6];
    const float* l0b = (const float*)d_in[7];
    const float* l1w = (const float*)d_in[8];
    const float* l1b = (const float*)d_in[9];
    const float* l2w = (const float*)d_in[10];
    const float* l2b = (const float*)d_in[11];
    const float* frq = (const float*)d_in[12];
    const float* fw  = (const float*)d_in[13];
    const float* mw  = (const float*)d_in[14];
    float* outp = (float*)d_out;

    // workspace carve-up (~113 MB)
    __bf16* krevb = (__bf16*)d_ws;                        // HD*KREV bf16 (0.5 MB)
    float*  qc    = (float*)((char*)d_ws + (size_t)HD * KREV * sizeof(__bf16));
    float*  kc    = qc + (size_t)BB * NH * HD * LSEQ;     // 16 MB each
    float*  vc    = kc + (size_t)BB * NH * HD * LSEQ;
    __bf16* kvb   = (__bf16*)(vc + (size_t)BB * NH * HD * LSEQ); // 64 MB

    hyena_filter_kernel<<<dim3(LSEQ / 128), dim3(128), 0, stream>>>(
        l0w, l0b, l1w, l1b, l2w, l2b, frq, fw, mw, krevb);

    const dim3 cg(LSEQ / 32, HD / 32, BB * NH);
    shortconv_kernel<<<cg, dim3(256), 0, stream>>>(qin, scw + 0 * CCH * 3, scb + 0 * CCH, qc);
    shortconv_kernel<<<cg, dim3(256), 0, stream>>>(kin, scw + 1 * CCH * 3, scb + 1 * CCH, kc);
    shortconv_kernel<<<cg, dim3(256), 0, stream>>>(vin, scw + 2 * CCH * 3, scb + 2 * CCH, vc);

    form_kv_kernel<<<dim3((HD * BB * NH * LSEQ) / 256), dim3(256), 0, stream>>>(vc, kc, kvb);

    // blocks sharing hd are adjacent in x -> per-hd 512KB kv slab stays
    // L2-resident across the whole triangle sweep (192MB L2, 23.3TB/s HBM)
    hyena_tconv_wmma_kernel<<<dim3(LSEQ / 64, HD), dim3(256), 0, stream>>>(
        krevb, kvb, qc, Dv, outp);
}
